// PointTextFusion_45698452029981
// MI455X (gfx1250) — matrix operand order repaired and checked by reading, not compiled, not run on hardware
//
#include <hip/hip_runtime.h>
#include <hip/hip_bf16.h>

// ---------------- problem constants ----------------
#define BB   4
#define NP   16384
#define LT   64
#define HH   256
#define SS   256
#define NHD  8
#define DPP  256
#define DTT  768
#define DVV  512
#define FDD  256
#define NROW (BB*NP)        // 65536 point rows
#define SPROW (BB*SS)       // 1024 superpoint rows
#define TROW (BB*LT)        // 256 text rows

typedef __attribute__((ext_vector_type(16))) __bf16 v16bf;
typedef __attribute__((ext_vector_type(8)))  float  v8f;
typedef __attribute__((ext_vector_type(4)))  unsigned int u32x4;
typedef __attribute__((ext_vector_type(8)))  int i32x8;
typedef __attribute__((ext_vector_type(4)))  int i32x4;

#if defined(__has_builtin)
#if __has_builtin(__builtin_amdgcn_tensor_load_to_lds)
#define USE_TDM 1
#endif
#if __has_builtin(__builtin_amdgcn_cvt_pk_bf16_f32)
#define HAVE_CVT_PK_BF16 1
#endif
#if __has_builtin(__builtin_amdgcn_sched_barrier)
#define SCHED_FENCE() __builtin_amdgcn_sched_barrier(0)
#endif
#endif
#ifndef USE_TDM
#define USE_TDM 0
#endif
#ifndef HAVE_CVT_PK_BF16
#define HAVE_CVT_PK_BF16 0
#endif
#ifndef SCHED_FENCE
#define SCHED_FENCE() ((void)0)
#endif

__device__ __forceinline__ unsigned int pack2bf(float a, float b) {
#if HAVE_CVT_PK_BF16
    typedef __attribute__((ext_vector_type(2))) __bf16 v2bf;
    union { v2bf v; unsigned int u; } c;
    c.v = __builtin_amdgcn_cvt_pk_bf16_f32(a, b);
    return c.u;
#else
    unsigned int ua = __float_as_uint(a), ub = __float_as_uint(b);
    ua += 0x7FFFu + ((ua >> 16) & 1u);
    ub += 0x7FFFu + ((ub >> 16) & 1u);
    return (ua >> 16) | (ub & 0xFFFF0000u);
#endif
}

#if USE_TDM
// Issue a TDM transfer of one contiguous 16KB B-slab (4096 dwords, 1 row)
// from global memory to LDS. D# layout per CDNA5 ISA 8.3/8.4.
__device__ __forceinline__ void tdm_load_slab(const void* gaddr, unsigned lds_off) {
    unsigned long long ga = (unsigned long long)gaddr;
    u32x4 g0;
    g0[0] = 1u;                                   // count=1, user mode
    g0[1] = lds_off;                              // lds_addr (bytes)
    g0[2] = (unsigned)ga;                         // global_addr[31:0]
    g0[3] = (unsigned)(ga >> 32) | 0x80000000u;   // global_addr[56:32] | type=2<<30
    i32x8 g1 = {0, 0, 0, 0, 0, 0, 0, 0};
    g1[0] = 0x00020000;          // data_size=2 (4 bytes); wg_mask=0; no pad/iter
    g1[1] = (int)(4096u << 16);  // tensor_dim0[15:0]=4096 at bits 63:48 of group
    g1[2] = 0x00010000;          // tensor_dim1=1 (bits 111:80)
    g1[3] = (int)(4096u << 16);  // tile_dim0=4096 (bits 127:112)
    g1[4] = 1;                   // tile_dim1=1
    g1[5] = 4096;                // tensor_dim0_stride=4096
    i32x4 g2 = {0, 0, 0, 0};
    i32x4 g3 = {0, 0, 0, 0};
#if __clang_major__ >= 23
    i32x8 g4 = {0, 0, 0, 0, 0, 0, 0, 0};
    __builtin_amdgcn_tensor_load_to_lds(g0, g1, g2, g3, g4, 0);
#else
    __builtin_amdgcn_tensor_load_to_lds(g0, g1, g2, g3, 0);
#endif
}
#endif

// ---------------- weight pre-pack into WMMA B-fragment layout ----------------
// W is (K, 256) row-major f32. frag = kt*16 + nt; per frag 32 lanes x 16 bf16.
// B element i of lane: K = kt*32 + i + 16*(lane/16), N = nt*16 + lane%16.
__global__ __launch_bounds__(256) void pack_w(const float* __restrict__ W, int Ktot,
                                              unsigned short* __restrict__ out) {
    int tid  = blockIdx.x * 256 + threadIdx.x;       // tid < Ktot*16
    int lane = tid & 31;
    int frag = tid >> 5;
    int kt = frag >> 4, nt = frag & 15;
    int half = lane >> 4, lp = lane & 15;
    int n = nt * 16 + lp;
    unsigned int u[8];
#pragma unroll
    for (int j = 0; j < 8; ++j) {
        int k = kt * 32 + 2 * j + 16 * half;
        u[j] = pack2bf(W[(size_t)k * 256 + n], W[(size_t)(k + 1) * 256 + n]);
    }
    uint4* dst = (uint4*)(out + ((size_t)frag * 32 + lane) * 16);
    dst[0] = make_uint4(u[0], u[1], u[2], u[3]);
    dst[1] = make_uint4(u[4], u[5], u[6], u[7]);
}

// ---------------- fused GEMM: Out = concat(A1,A2) @ Bp + bias [+ residual] ----------------
// N fixed = 256. Block = 256 threads = 8 waves; block computes 128 rows x 256 cols.
// Per k-step (32), a 16KB packed-B slab is staged in LDS (double-buffered via
// TDM when available). Inside the k-step, B fragments are consumed in ping-pong
// groups of 4 with sched_barriers pinning the shape: 8 ds_load_b128 for group
// g+1 issue first, then 4 back-to-back WMMAs for group g.
__global__ __launch_bounds__(256)
void gemm_bf16(const float* __restrict__ A1, int ldA1,
               const float* __restrict__ A2, int ldA2, int K1, int Ktot,
               const unsigned short* __restrict__ Bp,
               const float* __restrict__ bias,
               const float* residual, float* Out) {
    __shared__ uint4 ldsB[2048];  // 2 x 16KB slabs
    const int t = threadIdx.x;
    const int w = t >> 5, lane = t & 31;
    const int half = lane >> 4, lp = lane & 15;
    const int blockRow = blockIdx.x * 128;
    const int row = blockRow + w * 16 + lp;   // A row handled by this lane

    v8f zero = {};
    v8f acc[16];
#pragma unroll
    for (int i = 0; i < 16; ++i) acc[i] = zero;

    const int nKt = Ktot >> 5;

#if USE_TDM
    const unsigned ldsbase = (unsigned)(unsigned long long)(void*)&ldsB[0];
    if (w == 0) tdm_load_slab(Bp, ldsbase);
#else
    uint4 st[4];
    {
        const uint4* s0 = (const uint4*)Bp;
#pragma unroll
        for (int i = 0; i < 4; ++i) st[i] = s0[t + 256 * i];
#pragma unroll
        for (int i = 0; i < 4; ++i) ldsB[t + 256 * i] = st[i];
    }
#endif

    for (int kt = 0; kt < nKt; ++kt) {
        // A-fragment global loads for this k-step (independent of LDS traffic)
        int k0 = kt * 32;
        const float* Arow;
        int kin;
        if (k0 < K1) { Arow = A1 + (size_t)row * ldA1; kin = k0; }
        else         { Arow = A2 + (size_t)row * ldA2; kin = k0 - K1; }
        const float4* p0 = (const float4*)(Arow + kin + 16 * half);
        const float4* p1 = (const float4*)(Arow + kin + 16 * half + 8);
        float4 f0 = p0[0], f1 = p0[1], f2 = p1[0], f3 = p1[1];

#if USE_TDM
        if (w == 0) {
            if (kt + 1 < nKt) {
                tdm_load_slab(Bp + (size_t)(kt + 1) * 8192,
                              ldsbase + (unsigned)(((kt + 1) & 1) * 16384));
                __builtin_amdgcn_s_wait_tensorcnt(1);  // slab kt complete
            } else {
                __builtin_amdgcn_s_wait_tensorcnt(0);
            }
        }
#else
        if (kt + 1 < nKt) {
            const uint4* sn = (const uint4*)(Bp + (size_t)(kt + 1) * 8192);
#pragma unroll
            for (int i = 0; i < 4; ++i) st[i] = sn[t + 256 * i];
        }
#endif
        __syncthreads();  // slab kt visible to all waves

        union { unsigned int u[8]; v16bf v; } afr;
        afr.u[0] = pack2bf(f0.x, f0.y); afr.u[1] = pack2bf(f0.z, f0.w);
        afr.u[2] = pack2bf(f1.x, f1.y); afr.u[3] = pack2bf(f1.z, f1.w);
        afr.u[4] = pack2bf(f2.x, f2.y); afr.u[5] = pack2bf(f2.z, f2.w);
        afr.u[6] = pack2bf(f3.x, f3.y); afr.u[7] = pack2bf(f3.z, f3.w);

        const uint4* lb = ldsB + (kt & 1) * 1024;  // active slab
        union BF { uint4 q[2]; v16bf v; };
        BF bfA[4], bfB[4];  // ping-pong groups of 4 fragments
#pragma unroll
        for (int i = 0; i < 4; ++i) {
            bfA[i].q[0] = lb[(i * 32 + lane) * 2];
            bfA[i].q[1] = lb[(i * 32 + lane) * 2 + 1];
        }
        SCHED_FENCE();
#pragma unroll
        for (int g = 0; g < 4; ++g) {
            if ((g & 1) == 0) {
                if (g < 3) {
#pragma unroll
                    for (int i = 0; i < 4; ++i) {
                        int f = (g + 1) * 4 + i;
                        bfB[i].q[0] = lb[(f * 32 + lane) * 2];
                        bfB[i].q[1] = lb[(f * 32 + lane) * 2 + 1];
                    }
                }
                SCHED_FENCE();  // loads of group g+1 stay above WMMAs of group g
#pragma unroll
                for (int i = 0; i < 4; ++i) {
                    int nt = g * 4 + i;
                    acc[nt] = __builtin_amdgcn_wmma_f32_16x16x32_bf16(
                        false, afr.v, false, bfA[i].v, (short)0, acc[nt], false, false);
                }
                SCHED_FENCE();
            } else {
                if (g < 3) {
#pragma unroll
                    for (int i = 0; i < 4; ++i) {
                        int f = (g + 1) * 4 + i;
                        bfA[i].q[0] = lb[(f * 32 + lane) * 2];
                        bfA[i].q[1] = lb[(f * 32 + lane) * 2 + 1];
                    }
                }
                SCHED_FENCE();
#pragma unroll
                for (int i = 0; i < 4; ++i) {
                    int nt = g * 4 + i;
                    acc[nt] = __builtin_amdgcn_wmma_f32_16x16x32_bf16(
                        false, afr.v, false, bfB[i].v, (short)0, acc[nt], false, false);
                }
                SCHED_FENCE();
            }
        }

#if USE_TDM
        __syncthreads();  // all reads of slab kt done before TDM reuses its buffer
#else
        if (kt + 1 < nKt) {
            int d = ((kt + 1) & 1) * 1024;
#pragma unroll
            for (int i = 0; i < 4; ++i) ldsB[d + t + 256 * i] = st[i];
        }
#endif
    }

    // epilogue: bias (+ residual)
#pragma unroll
    for (int nt = 0; nt < 16; ++nt) {
        int col = nt * 16 + lp;
        float bv = bias[col];
#pragma unroll
        for (int j = 0; j < 8; ++j) {
            int r = blockRow + w * 16 + j + 8 * half;
            size_t idx = (size_t)r * 256 + col;
            float rv = residual ? residual[idx] : 0.0f;
            Out[idx] = acc[nt][j] + bv + rv;
        }
    }
}

// ---------------- in-place LayerNorm + ReLU over last dim (256) ----------------
__global__ __launch_bounds__(256)
void ln_relu(float* __restrict__ X, const float* __restrict__ g, const float* __restrict__ be) {
    int w = threadIdx.x >> 5, lane = threadIdx.x & 31;
    size_t row = (size_t)blockIdx.x * 8 + w;
    float* x = X + row * 256;
    float v[8], s = 0.0f, sq = 0.0f;
#pragma unroll
    for (int i = 0; i < 8; ++i) { v[i] = x[i * 32 + lane]; s += v[i]; sq += v[i] * v[i]; }
#pragma unroll
    for (int o = 16; o >= 1; o >>= 1) { s += __shfl_xor(s, o, 32); sq += __shfl_xor(sq, o, 32); }
    float mu  = s * (1.0f / 256.0f);
    float var = sq * (1.0f / 256.0f) - mu * mu;
    float rs  = rsqrtf(var + 1e-5f);
#pragma unroll
    for (int i = 0; i < 8; ++i) {
        int c = i * 32 + lane;
        x[c] = fmaxf(g[c] * (v[i] - mu) * rs + be[c], 0.0f);
    }
}

// ---------------- segment mean over superpoints ----------------
__global__ void zero_f32(float* p, int n) {
    int i = blockIdx.x * 256 + threadIdx.x;
    if (i < n) p[i] = 0.0f;
}

__global__ __launch_bounds__(256)
void seg_accum(const float* __restrict__ P, const int* __restrict__ sid,
               float* sums, float* cnt) {
    size_t gid = (size_t)blockIdx.x * 256 + threadIdx.x;  // row*64 + c4
    int c4 = (int)(gid & 63);
    size_t rowg = gid >> 6;
    int b = (int)(rowg >> 14);                            // Np = 16384
    int seg = b * SS + sid[rowg];
    float4 v = ((const float4*)(P + rowg * 256))[c4];
    float* d = sums + (size_t)seg * 256 + c4 * 4;
    atomicAdd(d + 0, v.x); atomicAdd(d + 1, v.y);
    atomicAdd(d + 2, v.z); atomicAdd(d + 3, v.w);
    if (c4 == 0) atomicAdd(cnt + seg, 1.0f);
}

__global__ void seg_mean(const float* __restrict__ sums, const float* __restrict__ cnt,
                         float* __restrict__ sp) {
    int i = blockIdx.x * 256 + threadIdx.x;  // 1024*256 elements
    int row = i >> 8;
    sp[i] = sums[i] / fmaxf(cnt[row], 1.0f);
}

// ---------------- superpoint->text multi-head attention (S=256, Lt=64, d=32) ----------------
__global__ __launch_bounds__(256)
void attention(const float* __restrict__ q, const float* __restrict__ k,
               const float* __restrict__ v, const unsigned char* __restrict__ mask,
               float* __restrict__ o) {
    __shared__ float sK[64 * 32], sV[64 * 32];
    int b = blockIdx.x >> 3, h = blockIdx.x & 7;
    int t = threadIdx.x;  // query index s
    for (int i = t; i < 64 * 32; i += 256) {
        int kk = i >> 5, d = i & 31;
        size_t off = ((size_t)b * LT + kk) * 256 + h * 32 + d;
        sK[i] = k[off];
        sV[i] = v[off];
    }
    __syncthreads();
    float qr[32];
    const float* qp = q + ((size_t)b * SS + t) * 256 + h * 32;
#pragma unroll
    for (int d = 0; d < 32; ++d) qr[d] = qp[d];
    float sc[64], mx = -1e30f;
    for (int kk = 0; kk < 64; ++kk) {
        float s = 0.0f;
#pragma unroll
        for (int d = 0; d < 32; ++d) s += qr[d] * sK[kk * 32 + d];
        s *= 0.17677669529663687f;  // 1/sqrt(32)
        if (!mask[b * LT + kk]) s = -1e9f;
        sc[kk] = s;
        mx = fmaxf(mx, s);
    }
    float den = 0.0f;
    for (int kk = 0; kk < 64; ++kk) { sc[kk] = __expf(sc[kk] - mx); den += sc[kk]; }
    float inv = 1.0f / den;
    float* op = o + ((size_t)b * SS + t) * 256 + h * 32;
#pragma unroll
    for (int d = 0; d < 32; ++d) {
        float s = 0.0f;
        for (int kk = 0; kk < 64; ++kk) s += sc[kk] * sV[kk * 32 + d];
        op[d] = s * inv;
    }
}

// ---------------- gather enhanced superpoint features back to points ----------------
__global__ __launch_bounds__(256)
void gather_E(const float* __restrict__ enh, const int* __restrict__ sid,
              float* __restrict__ E) {
    size_t gid = (size_t)blockIdx.x * 256 + threadIdx.x;  // row*64 + c4
    int c4 = (int)(gid & 63);
    size_t rowg = gid >> 6;
    int b = (int)(rowg >> 14);
    int s = sid[rowg];
    ((float4*)(E + rowg * 256))[c4] =
        ((const float4*)(enh + ((size_t)(b * SS + s)) * 256))[c4];
}

// ---------------- host-side orchestration ----------------
extern "C" void kernel_launch(void* const* d_in, const int* in_sizes, int n_in,
                              void* d_out, int out_size, void* d_ws, size_t ws_size,
                              hipStream_t stream) {
    (void)in_sizes; (void)n_in; (void)out_size; (void)ws_size;
    const float* PF   = (const float*)d_in[0];
    const float* TF   = (const float*)d_in[1];
    const float* ZPV  = (const float*)d_in[2];
    const float* ZTV  = (const float*)d_in[3];
    const int*   SID  = (const int*)d_in[4];
    const unsigned char* MASK = (const unsigned char*)d_in[5];
    // params (dict order)
    const float* Wp  = (const float*)d_in[6];  const float* bp  = (const float*)d_in[7];
    const float* gp  = (const float*)d_in[8];  const float* bep = (const float*)d_in[9];
    const float* Wt  = (const float*)d_in[10]; const float* bt  = (const float*)d_in[11];
    const float* gt  = (const float*)d_in[12]; const float* bet = (const float*)d_in[13];
    const float* Wv1 = (const float*)d_in[14]; const float* bv1 = (const float*)d_in[15];
    const float* gv  = (const float*)d_in[16]; const float* bev = (const float*)d_in[17];
    const float* Wv2 = (const float*)d_in[18]; const float* bv2 = (const float*)d_in[19];
    const float* Ws1 = (const float*)d_in[20]; const float* bs1 = (const float*)d_in[21];
    const float* gs  = (const float*)d_in[22]; const float* bes = (const float*)d_in[23];
    const float* Ws2 = (const float*)d_in[24]; const float* bs2 = (const float*)d_in[25];
    const float* Wq  = (const float*)d_in[26]; const float* bq  = (const float*)d_in[27];
    const float* Wk  = (const float*)d_in[28]; const float* bk  = (const float*)d_in[29];
    const float* Wva = (const float*)d_in[30]; const float* bva = (const float*)d_in[31];
    const float* Wo  = (const float*)d_in[32]; const float* bo  = (const float*)d_in[33];
    const float* Wi  = (const float*)d_in[34]; const float* bi  = (const float*)d_in[35];
    const float* gi  = (const float*)d_in[36]; const float* bei = (const float*)d_in[37];
    const float* Wout= (const float*)d_in[38]; const float* bout= (const float*)d_in[39];

    // workspace carve-up (f32)
    float* P    = (float*)d_ws;                  // 65536x256
    float* VhE  = P    + (size_t)NROW * HH;      // 65536x256 (Vh, later E)
    float* Zb   = VhE  + (size_t)NROW * HH;      // 65536x256
    float* Tb   = Zb   + (size_t)NROW * HH;      // 256x256
    float* sums = Tb   + (size_t)TROW * HH;      // 1024x256
    float* cnt  = sums + (size_t)SPROW * HH;     // 1024
    float* sp   = cnt  + SPROW;                  // 1024x256
    float* hb   = sp   + (size_t)SPROW * HH;     // 1024x256
    float* spm  = hb   + (size_t)SPROW * HH;     // 1024x256 (sp_sem)
    float* qb   = spm  + (size_t)SPROW * HH;     // 1024x256
    float* kb   = qb   + (size_t)SPROW * HH;     // 256x256
    float* vb   = kb   + (size_t)TROW * HH;      // 256x256
    float* ob   = vb   + (size_t)TROW * HH;      // 1024x256
    float* enh  = ob   + (size_t)SPROW * HH;     // 1024x256
    unsigned short* pk = (unsigned short*)(enh + (size_t)SPROW * HH);
    unsigned short* pWp  = pk;
    unsigned short* pWt  = pWp  + 65536;
    unsigned short* pWv1 = pWt  + 196608;
    unsigned short* pWv2 = pWv1 + 262144;
    unsigned short* pWs1 = pWv2 + 65536;
    unsigned short* pWs2 = pWs1 + 65536;
    unsigned short* pWq  = pWs2 + 65536;
    unsigned short* pWk  = pWq  + 65536;
    unsigned short* pWva = pWk  + 65536;
    unsigned short* pWo  = pWva + 65536;
    unsigned short* pWi  = pWo  + 65536;
    unsigned short* pWout= pWi  + 131072;

    auto PK = [&](const float* W, int K, unsigned short* dst) {
        pack_w<<<dim3((K * 16) / 256), dim3(256), 0, stream>>>(W, K, dst);
    };
    auto GM = [&](const float* A1, int ldA1, const float* A2, int ldA2, int K1, int K,
                  const unsigned short* Bp, const float* bias, const float* res,
                  float* out, int M) {
        gemm_bf16<<<dim3(M / 128), dim3(256), 0, stream>>>(A1, ldA1, A2, ldA2, K1, K,
                                                           Bp, bias, res, out);
    };
    auto LN = [&](float* X, const float* g, const float* be, int rows) {
        ln_relu<<<dim3(rows / 8), dim3(256), 0, stream>>>(X, g, be);
    };

    // 0) pack all weights to bf16 fragments
    PK(Wp, DPP, pWp);    PK(Wt, DTT, pWt);    PK(Wv1, 2 * DVV, pWv1);
    PK(Wv2, HH, pWv2);   PK(Ws1, HH, pWs1);   PK(Ws2, HH, pWs2);
    PK(Wq, HH, pWq);     PK(Wk, HH, pWk);     PK(Wva, HH, pWva);
    PK(Wo, HH, pWo);     PK(Wi, 2 * HH, pWi); PK(Wout, HH, pWout);

    // 1) P = relu(LN(PF @ Wp + bp))
    GM(PF, DPP, nullptr, 0, DPP, DPP, pWp, bp, nullptr, P, NROW);
    LN(P, gp, bep, NROW);
    // 2) T = relu(LN(TF @ Wt + bt))
    GM(TF, DTT, nullptr, 0, DTT, DTT, pWt, bt, nullptr, Tb, TROW);
    LN(Tb, gt, bet, TROW);
    // 3) Vh = relu(LN(concat(Z_PV, Z_TV) @ Wv1 + bv1))
    GM(ZPV, DVV, ZTV, DVV, DVV, 2 * DVV, pWv1, bv1, nullptr, VhE, NROW);
    LN(VhE, gv, bev, NROW);
    // 4) P = P + Vh @ Wv2 + bv2
    GM(VhE, HH, nullptr, 0, HH, HH, pWv2, bv2, P, P, NROW);
    // 5) superpoint mean pooling
    zero_f32<<<dim3((SPROW * HH + SPROW + 255) / 256), dim3(256), 0, stream>>>(
        sums, SPROW * HH + SPROW);
    seg_accum<<<dim3(NROW * 64 / 256), dim3(256), 0, stream>>>(P, SID, sums, cnt);
    seg_mean<<<dim3(SPROW * HH / 256), dim3(256), 0, stream>>>(sums, cnt, sp);
    // 6) h = relu(LN(sp @ Ws1 + bs1)); sp_sem = h @ Ws2 + bs2
    GM(sp, HH, nullptr, 0, HH, HH, pWs1, bs1, nullptr, hb, SPROW);
    LN(hb, gs, bes, SPROW);
    GM(hb, HH, nullptr, 0, HH, HH, pWs2, bs2, nullptr, spm, SPROW);
    // 7) q/k/v projections + attention
    GM(spm, HH, nullptr, 0, HH, HH, pWq, bq, nullptr, qb, SPROW);
    GM(Tb, HH, nullptr, 0, HH, HH, pWk, bk, nullptr, kb, TROW);
    GM(Tb, HH, nullptr, 0, HH, HH, pWva, bva, nullptr, vb, TROW);
    attention<<<dim3(BB * NHD), dim3(256), 0, stream>>>(qb, kb, vb, MASK, ob);
    // 8) enhanced = sp_sem + o @ Wo + bo
    GM(ob, HH, nullptr, 0, HH, HH, pWo, bo, spm, enh, SPROW);
    // 9) E = gather(enhanced, sid)  (reuse Vh buffer)
    gather_E<<<dim3(NROW * 64 / 256), dim3(256), 0, stream>>>(enh, SID, VhE);
    // 10) Z = relu(LN(concat(P, E) @ Wi + bi))
    GM(P, HH, VhE, HH, HH, 2 * HH, pWi, bi, nullptr, Zb, NROW);
    LN(Zb, gi, bei, NROW);
    // 11) out = Z @ Wout + bout
    GM(Zb, HH, nullptr, 0, HH, HH, pWout, bout, nullptr, (float*)d_out, NROW);
}
